// Decoder_86912958202171
// MI455X (gfx1250) — matrix operand order repaired and checked by reading, not compiled, not run on hardware
//
#include <hip/hip_runtime.h>
#include <math.h>

typedef __attribute__((ext_vector_type(16))) _Float16 v16h;
typedef __attribute__((ext_vector_type(8)))  _Float16 v8h;
typedef __attribute__((ext_vector_type(4)))  _Float16 v4h;
typedef __attribute__((ext_vector_type(2)))  __fp16   fp16x2;  // cvt_pkrtz return type
typedef __attribute__((ext_vector_type(8)))  float    v8f;

static constexpr int B_  = 32;
static constexpr int S_  = 196;
static constexpr int T_  = 128;
static constexpr int V1  = 10001;   // V + 1
static constexpr int V1P = 10016;   // padded logit ld (mult of 32)
static constexpr int CI  = 2048;    // img feature channels
static constexpr int D_  = 512;
static constexpr int H_  = 8;
static constexpr int NBLK = 6;
static constexpr int SCLD = 224;    // padded row pitch for cross scores
static constexpr int BT  = B_ * T_;     // 4096
static constexpr int BS  = B_ * S_;     // 6272

// ---------------------------------------------------------------------------
// Generic batched GEMM: C = act(A @ B + bias) (+ residual), f32 in/out,
// f16 WMMA compute (16x16x32, f32 accumulate). Block = 8 waves, 128x64 tile.
// M must be a multiple of 128 (true for all launches below).
// Fast path: double-buffered LDS, prefetch-next-tile pipelining, 1 barrier/iter.
// ---------------------------------------------------------------------------
#define LDSA 40   // LDS row stride in halves (80B rows: 16B-aligned chunks)
#define LDSB 40

__global__ __launch_bounds__(256)
void k_gemm(const float* __restrict__ A, const float* __restrict__ Bm,
            const float* __restrict__ bias, const float* __restrict__ Res,
            float* __restrict__ C,
            int M, int N, int K, int lda, int ldb, int ldc,
            long long sA, long long sB, long long sC,
            int transB, int act, int hasBias, int hasRes)
{
  __shared__ _Float16 sA2[2][128 * LDSA];
  __shared__ _Float16 sB2[2][64 * LDSB];

  const int z = blockIdx.z;
  A += (long long)z * sA;
  Bm += (long long)z * sB;
  C += (long long)z * sC;
  const float* R = hasRes ? (Res + (long long)z * sC) : nullptr;

  const int mBase = blockIdx.y * 128;
  const int nBase = blockIdx.x * 64;
  const int tid  = threadIdx.x;
  const int lane = tid & 31;
  const int wave = tid >> 5;
  const int wm = wave & 3;   // 0..3  (M sub-tile)
  const int wn = wave >> 2;  // 0..1  (N sub-tile)

  const int mrow = lane & 15;
  const int kbA = (lane >> 4) * 8;
  const int kbB = (lane >> 4) * 8;

  v8f acc[2][2] = {};
  const int kTiles = (K + 31) >> 5;

  const bool fastAll = ((lda & 3) == 0) && ((ldb & 3) == 0) &&
                       ((K & 31) == 0) && (nBase + 64 <= N);

  if (fastAll) {
    float4 ra[4], rb[2];

    auto fetch = [&](int k0) {
      #pragma unroll
      for (int i = 0; i < 4; ++i) {
        int idx = tid + i * 256;        // 0..1023
        int r  = idx >> 3;              // 0..127
        int c4 = (idx & 7) << 2;        // 0..28
        ra[i] = *(const float4*)(A + (long long)(mBase + r) * lda + k0 + c4);
      }
      if (transB) {
        #pragma unroll
        for (int i = 0; i < 2; ++i) {
          int idx = tid + i * 256;      // 0..511
          int n  = idx >> 3;            // 0..63
          int c4 = (idx & 7) << 2;
          rb[i] = *(const float4*)(Bm + (long long)(nBase + n) * ldb + k0 + c4);
        }
      } else {
        #pragma unroll
        for (int i = 0; i < 2; ++i) {
          int idx = tid + i * 256;
          int k  = idx >> 4;            // 0..31
          int n4 = (idx & 15) << 2;     // 0..60
          rb[i] = *(const float4*)(Bm + (long long)(k0 + k) * ldb + nBase + n4);
        }
      }
    };

    auto stage = [&](int buf) {
      _Float16* sa = sA2[buf];
      _Float16* sb = sB2[buf];
      #pragma unroll
      for (int i = 0; i < 4; ++i) {
        int idx = tid + i * 256;
        int r  = idx >> 3;
        int c4 = (idx & 7) << 2;
        union { v4h v4; fp16x2 h2[2]; } u;
        u.h2[0] = __builtin_amdgcn_cvt_pkrtz(ra[i].x, ra[i].y);
        u.h2[1] = __builtin_amdgcn_cvt_pkrtz(ra[i].z, ra[i].w);
        *(v4h*)&sa[r * LDSA + c4] = u.v4;
      }
      if (transB) {
        #pragma unroll
        for (int i = 0; i < 2; ++i) {
          int idx = tid + i * 256;
          int n  = idx >> 3;
          int c4 = (idx & 7) << 2;
          union { v4h v4; fp16x2 h2[2]; } u;
          u.h2[0] = __builtin_amdgcn_cvt_pkrtz(rb[i].x, rb[i].y);
          u.h2[1] = __builtin_amdgcn_cvt_pkrtz(rb[i].z, rb[i].w);
          *(v4h*)&sb[n * LDSB + c4] = u.v4;
        }
      } else {
        #pragma unroll
        for (int i = 0; i < 2; ++i) {
          int idx = tid + i * 256;
          int k  = idx >> 4;
          int n4 = (idx & 15) << 2;
          sb[(n4 + 0) * LDSB + k] = (_Float16)rb[i].x;
          sb[(n4 + 1) * LDSB + k] = (_Float16)rb[i].y;
          sb[(n4 + 2) * LDSB + k] = (_Float16)rb[i].z;
          sb[(n4 + 3) * LDSB + k] = (_Float16)rb[i].w;
        }
      }
    };

    fetch(0);
    stage(0);
    __syncthreads();

    for (int kt = 0; kt < kTiles; ++kt) {
      const int cur = kt & 1;
      const bool more = (kt + 1 < kTiles);
      if (more) fetch((kt + 1) << 5);     // issue next-tile global loads early

      const _Float16* sa = sA2[cur];
      const _Float16* sb = sB2[cur];
      v16h aF[2], bF[2];
      #pragma unroll
      for (int t = 0; t < 2; ++t) {
        int r = wm * 32 + t * 16 + mrow;
        union { v16h v; v8h h[2]; } u;
        u.h[0] = *(const v8h*)&sa[r * LDSA + kbA];
        u.h[1] = *(const v8h*)&sa[r * LDSA + kbA + 16];
        aF[t] = u.v;
      }
      #pragma unroll
      for (int t = 0; t < 2; ++t) {
        int n = wn * 32 + t * 16 + mrow;
        union { v16h v; v8h h[2]; } u;
        u.h[0] = *(const v8h*)&sb[n * LDSB + kbB];
        u.h[1] = *(const v8h*)&sb[n * LDSB + kbB + 16];
        bF[t] = u.v;
      }
      #pragma unroll
      for (int ti = 0; ti < 2; ++ti)
        #pragma unroll
        for (int tj = 0; tj < 2; ++tj)
          acc[ti][tj] = __builtin_amdgcn_wmma_f32_16x16x32_f16(
              false, aF[ti], false, bF[tj], (short)0, acc[ti][tj], false, false);

      if (more) stage(cur ^ 1);           // waits loads, fills other buffer
      __syncthreads();
    }
  } else {
    // ---- generic guarded path (tail blocks only) ----
    for (int kt = 0; kt < kTiles; ++kt) {
      const int k0 = kt << 5;
      #pragma unroll
      for (int i = 0; i < 16; ++i) {
        int idx = tid + i * 256;
        int r = idx >> 5, c = idx & 31;
        int gk = k0 + c;
        float v = (gk < K) ? A[(long long)(mBase + r) * lda + gk] : 0.0f;
        sA2[0][r * LDSA + c] = (_Float16)v;
      }
      #pragma unroll
      for (int i = 0; i < 8; ++i) {
        int idx = tid + i * 256;
        int n = idx >> 5, c = idx & 31;
        int gn = nBase + n, gk = k0 + c;
        float v = 0.0f;
        if (gn < N && gk < K)
          v = transB ? Bm[(long long)gn * ldb + gk] : Bm[(long long)gk * ldb + gn];
        sB2[0][n * LDSB + c] = (_Float16)v;
      }
      __syncthreads();

      v16h aF[2], bF[2];
      #pragma unroll
      for (int t = 0; t < 2; ++t) {
        int r = wm * 32 + t * 16 + mrow;
        union { v16h v; v8h h[2]; } u;
        u.h[0] = *(const v8h*)&sA2[0][r * LDSA + kbA];
        u.h[1] = *(const v8h*)&sA2[0][r * LDSA + kbA + 16];
        aF[t] = u.v;
      }
      #pragma unroll
      for (int t = 0; t < 2; ++t) {
        int n = wn * 32 + t * 16 + mrow;
        union { v16h v; v8h h[2]; } u;
        u.h[0] = *(const v8h*)&sB2[0][n * LDSB + kbB];
        u.h[1] = *(const v8h*)&sB2[0][n * LDSB + kbB + 16];
        bF[t] = u.v;
      }
      #pragma unroll
      for (int ti = 0; ti < 2; ++ti)
        #pragma unroll
        for (int tj = 0; tj < 2; ++tj)
          acc[ti][tj] = __builtin_amdgcn_wmma_f32_16x16x32_f16(
              false, aF[ti], false, bF[tj], (short)0, acc[ti][tj], false, false);
      __syncthreads();
    }
  }

  // ---- epilogue: D layout — col = lane&15, row(local) = vgpr + 8*(lane>=16)
  const int nl = lane & 15;
  const int rb_ = (lane >> 4) * 8;
  #pragma unroll
  for (int tj = 0; tj < 2; ++tj) {
    int col = nBase + wn * 32 + tj * 16 + nl;
    if (col >= N) continue;
    float bv = hasBias ? bias[col] : 0.0f;
    #pragma unroll
    for (int ti = 0; ti < 2; ++ti) {
      int row0 = mBase + wm * 32 + ti * 16 + rb_;
      #pragma unroll
      for (int r = 0; r < 8; ++r) {
        float v = acc[ti][tj][r] + bv;
        if (act == 1) v = v > 0.f ? v : 0.f;            // relu
        else if (act == 2) v = v > 0.f ? v : 0.01f * v; // leaky_relu
        if (hasRes) v += R[(long long)(row0 + r) * ldc + col];
        C[(long long)(row0 + r) * ldc + col] = v;
      }
    }
  }
}

// ---------------------------------------------------------------------------
// Embedding gather + sinusoidal positional encoding; also emits tpad.
// ---------------------------------------------------------------------------
__global__ __launch_bounds__(512)
void k_embed(const int* __restrict__ tgt, const float* __restrict__ emb,
             float* __restrict__ X, float* __restrict__ tpad)
{
  int bt = blockIdx.x;
  int t = bt & (T_ - 1);
  int d = threadIdx.x;
  int tok = tgt[bt];
  int i = d >> 1;
  const float LOG1E4 = 9.210340371976184f;   // ln(10000)
  float div = expf(-((2.0f * (float)i) / (float)D_) * LOG1E4);
  float ang = (float)t * div;
  float pe = (d & 1) ? cosf(ang) : sinf(ang);
  X[(long long)bt * D_ + d] = emb[(long long)tok * D_ + d] + pe;
  if (d == 0) tpad[bt] = (tok == 0) ? 0.0f : 1.0f;
}

// ---------------------------------------------------------------------------
// Masked softmax over one score row; zeros the [L, ld) pad region so the
// following A@V GEMM can run with K = ld (multiple of 32).
// ---------------------------------------------------------------------------
__global__ __launch_bounds__(256)
void k_softmax(float* __restrict__ Sc, const float* __restrict__ tpad,
               const float* __restrict__ srcpad,
               int L, int ld, int causal, float scale)
{
  __shared__ float red[256];
  int row = blockIdx.x;          // b*T + t
  int b = row >> 7;              // T_ == 128
  int t = row & (T_ - 1);
  float* p = Sc + (long long)row * ld;
  int j = threadIdx.x;

  float qv = tpad[b * T_ + t];
  float val = -3.0e38f;
  if (j < L) {
    float kv = causal ? ((j <= t) ? tpad[b * T_ + j] : 0.0f)
                      : srcpad[b * L + j];
    bool keep = (qv != 0.0f) && (kv != 0.0f);
    val = keep ? p[j] * scale : -1e32f;
  }
  red[j] = val;
  __syncthreads();
  for (int s = 128; s > 0; s >>= 1) {
    if (j < s) red[j] = fmaxf(red[j], red[j + s]);
    __syncthreads();
  }
  float mx = red[0];
  __syncthreads();
  float e = (j < L) ? expf(val - mx) : 0.0f;
  red[j] = e;
  __syncthreads();
  for (int s = 128; s > 0; s >>= 1) {
    if (j < s) red[j] += red[j + s];
    __syncthreads();
  }
  float sum = red[0];
  if (j < L) p[j] = e / sum;
  else if (j < ld) p[j] = 0.0f;
}

// ---------------------------------------------------------------------------
// Fused residual + LayerNorm (in-place on X): X = LN(X + R) * g + b. D = 512.
// ---------------------------------------------------------------------------
__global__ __launch_bounds__(256)
void k_ln_res(float* __restrict__ X, const float* __restrict__ Rr,
              const float* __restrict__ g, const float* __restrict__ bt)
{
  __shared__ float red[256];
  long long row = blockIdx.x;
  float* x = X + row * D_;
  const float* r = Rr + row * D_;
  int tid = threadIdx.x;
  float v0 = x[tid] + r[tid];
  float v1 = x[tid + 256] + r[tid + 256];
  red[tid] = v0 + v1;
  __syncthreads();
  for (int s = 128; s > 0; s >>= 1) {
    if (tid < s) red[tid] += red[tid + s];
    __syncthreads();
  }
  float mean = red[0] * (1.0f / D_);
  __syncthreads();
  float d0 = v0 - mean, d1 = v1 - mean;
  red[tid] = d0 * d0 + d1 * d1;
  __syncthreads();
  for (int s = 128; s > 0; s >>= 1) {
    if (tid < s) red[tid] += red[tid + s];
    __syncthreads();
  }
  float inv = rsqrtf(red[0] * (1.0f / D_) + 1e-5f);
  x[tid]       = d0 * inv * g[tid]       + bt[tid];
  x[tid + 256] = d1 * inv * g[tid + 256] + bt[tid + 256];
}

// ---------------------------------------------------------------------------
// Copy block-0 cross-attention probs (H,B,T,SCLD) -> output (B,H,T,S).
// ---------------------------------------------------------------------------
__global__ __launch_bounds__(256)
void k_att_copy(const float* __restrict__ scc, float* __restrict__ out)
{
  long long i = (long long)blockIdx.x * 256 + threadIdx.x;
  const long long tot = (long long)B_ * H_ * T_ * S_;
  if (i >= tot) return;
  int s = (int)(i % S_); long long r = i / S_;
  int t = (int)(r % T_); r /= T_;
  int h = (int)(r % H_);
  int b = (int)(r / H_);
  out[i] = scc[(((long long)(h * B_ + b) * T_ + t) * SCLD) + s];
}

// ---------------------------------------------------------------------------
// Pad-copy w_out (D x V1, ld=V1) -> wpad (D x V1P, zeros in pad columns).
// ---------------------------------------------------------------------------
__global__ __launch_bounds__(256)
void k_padw(const float* __restrict__ w, float* __restrict__ wp)
{
  long long i = (long long)blockIdx.x * 256 + threadIdx.x;
  const long long tot = (long long)D_ * V1P;
  if (i >= tot) return;
  int c = (int)(i % V1P);
  long long r = i / V1P;
  wp[i] = (c < V1) ? w[r * V1 + c] : 0.0f;
}

// ---------------------------------------------------------------------------

extern "C" void kernel_launch(void* const* d_in, const int* in_sizes, int n_in,
                              void* d_out, int out_size, void* d_ws, size_t ws_size,
                              hipStream_t stream) {
  if (n_in < 3 + 9 + NBLK * 26) return;

  int idx = 0;
  const float* enc_src = (const float*)d_in[idx++];   // (B,S,CI)
  const float* src_pad = (const float*)d_in[idx++];   // (B,S)
  const int*   target  = (const int*)d_in[idx++];     // (B,T)
  const float* w_img  = (const float*)d_in[idx++];
  const float* b_img  = (const float*)d_in[idx++];
  const float* res_w1 = (const float*)d_in[idx++];
  const float* res_b1 = (const float*)d_in[idx++];
  const float* res_w2 = (const float*)d_in[idx++];
  const float* res_b2 = (const float*)d_in[idx++];
  const float* emb    = (const float*)d_in[idx++];
  const float* w_out  = (const float*)d_in[idx++];
  const float* b_out  = (const float*)d_in[idx++];

  struct Blk {
    const float *wq,*bq,*wk,*bk,*wv,*bv,*wo,*bo;          // attn1
    const float *wq2,*bq2,*wk2,*bk2,*wv2,*bv2,*wo2,*bo2;  // attn2
    const float *ln1g,*ln1b,*ln2g,*ln2b,*ln3g,*ln3b;
    const float *fw1,*fb1,*fw2,*fb2;
  } blk[NBLK];
  for (int i = 0; i < NBLK; ++i) {
    const float** f = (const float**)&blk[i];
    for (int j = 0; j < 26; ++j) f[j] = (const float*)d_in[idx++];
  }

  // workspace layout (floats)
  const long long o_x      = 0;                              // BT*D
  const long long o_tpad   = o_x      + (long long)BT * D_;  // B*T
  const long long o_enc    = o_tpad   + 8192;                // BS*D
  const long long o_bufA   = o_enc    + (long long)BS * D_;  // per-head Q / tmp
  const long long o_bufB   = o_bufA   + (long long)BS * D_;  // per-head K
  const long long o_bufV   = o_bufB   + (long long)BS * D_;  // per-head V
  const long long o_concat = o_bufV   + (long long)BS * D_;  // BT * H*D
  const long long o_scs    = o_concat + (long long)BT * H_ * D_;      // H*B*T*T
  const long long o_scc    = o_scs    + (long long)H_ * B_ * T_ * T_; // H*B*T*SCLD
  const long long o_hid    = o_scc    + (long long)H_ * B_ * T_ * SCLD;
  const long long o_tmp    = o_hid    + (long long)BT * 4 * D_;
  const long long o_wpad   = o_tmp    + (long long)BT * D_;  // D x V1P
  const long long o_end    = o_wpad   + (long long)D_ * V1P;
  if (ws_size < (size_t)o_end * sizeof(float)) return;

  float* ws = (float*)d_ws;
  float* x      = ws + o_x;
  float* tpad   = ws + o_tpad;
  float* enc    = ws + o_enc;
  float* bufA   = ws + o_bufA;
  float* bufB   = ws + o_bufB;
  float* bufV   = ws + o_bufV;
  float* concat = ws + o_concat;
  float* scs    = ws + o_scs;
  float* scc    = ws + o_scc;
  float* hid    = ws + o_hid;
  float* tmp    = ws + o_tmp;
  float* wpad   = ws + o_wpad;
  float* y_out   = (float*)d_out;
  float* att_out = y_out + (long long)B_ * T_ * V1;

  auto gemm = [&](const float* A, const float* Bm, const float* bias,
                  const float* Res, float* C,
                  int M, int N, int K, int lda, int ldb, int ldc,
                  long long sA, long long sB, long long sC, int nz,
                  int transB, int act) {
    dim3 g((N + 63) / 64, (M + 127) / 128, nz);
    k_gemm<<<g, 256, 0, stream>>>(A, Bm, bias, Res, C, M, N, K, lda, ldb, ldc,
                                  sA, sB, sC, transB, act,
                                  bias != nullptr, Res != nullptr);
  };

  const float scale = 1.0f / sqrtf((float)D_);
  const long long DD = (long long)D_ * D_;

  // embedding + positional encoding + tpad; pad-copy w_out
  k_embed<<<BT, 512, 0, stream>>>(target, emb, x, tpad);
  {
    long long tot = (long long)D_ * V1P;
    k_padw<<<(int)((tot + 255) / 256), 256, 0, stream>>>(w_out, wpad);
  }

  // image feature mapping + residual block
  gemm(enc_src, w_img, b_img, nullptr, enc, BS, D_, CI, CI, D_, D_, 0,0,0,1, 0, 2);
  gemm(enc, res_w1, res_b1, nullptr, bufA, BS, D_, D_, D_, D_, D_, 0,0,0,1, 0, 2);
  gemm(bufA, res_w2, res_b2, enc /*residual*/, enc, BS, D_, D_, D_, D_, D_, 0,0,0,1, 0, 0);

  for (int L = 0; L < NBLK; ++L) {
    Blk& p = blk[L];

    // ---- self attention (per head) ----
    for (int h = 0; h < H_; ++h) {
      gemm(x, p.wq + h*DD, p.bq + h*D_, nullptr, bufA, BT, D_, D_, D_, D_, D_, 0,0,0,1, 0, 0);
      gemm(x, p.wk + h*DD, p.bk + h*D_, nullptr, bufB, BT, D_, D_, D_, D_, D_, 0,0,0,1, 0, 0);
      gemm(x, p.wv + h*DD, p.bv + h*D_, nullptr, bufV, BT, D_, D_, D_, D_, D_, 0,0,0,1, 0, 0);
      float* sc = scs + (long long)h * B_ * T_ * T_;
      gemm(bufA, bufB, nullptr, nullptr, sc, T_, T_, D_, D_, D_, T_,
           (long long)T_*D_, (long long)T_*D_, (long long)T_*T_, B_, 1, 0);
      k_softmax<<<BT, 256, 0, stream>>>(sc, tpad, nullptr, T_, T_, 1, scale);
      gemm(sc, bufV, nullptr, nullptr, concat + h*D_, T_, D_, T_, T_, D_, H_*D_,
           (long long)T_*T_, (long long)T_*D_, (long long)T_*H_*D_, B_, 0, 0);
    }
    gemm(concat, p.wo, p.bo, nullptr, tmp, BT, D_, H_*D_, H_*D_, D_, D_, 0,0,0,1, 0, 0);
    k_ln_res<<<BT, 256, 0, stream>>>(x, tmp, p.ln1g, p.ln1b);

    // ---- cross attention (per head) ----
    for (int h = 0; h < H_; ++h) {
      gemm(x,   p.wq2 + h*DD, p.bq2 + h*D_, nullptr, bufA, BT, D_, D_, D_, D_, D_, 0,0,0,1, 0, 0);
      gemm(enc, p.wk2 + h*DD, p.bk2 + h*D_, nullptr, bufB, BS, D_, D_, D_, D_, D_, 0,0,0,1, 0, 0);
      gemm(enc, p.wv2 + h*DD, p.bv2 + h*D_, nullptr, bufV, BS, D_, D_, D_, D_, D_, 0,0,0,1, 0, 0);
      float* sc = scc + (long long)h * B_ * T_ * SCLD;
      gemm(bufA, bufB, nullptr, nullptr, sc, T_, S_, D_, D_, D_, SCLD,
           (long long)T_*D_, (long long)S_*D_, (long long)T_*SCLD, B_, 1, 0);
      k_softmax<<<BT, 256, 0, stream>>>(sc, tpad, src_pad, S_, SCLD, 0, scale);
      // K = SCLD (pad columns are exact zeros after softmax)
      gemm(sc, bufV, nullptr, nullptr, concat + h*D_, T_, D_, SCLD, SCLD, D_, H_*D_,
           (long long)T_*SCLD, (long long)S_*D_, (long long)T_*H_*D_, B_, 0, 0);
    }
    if (L == 0) {
      long long tot = (long long)B_ * H_ * T_ * S_;
      k_att_copy<<<(int)((tot + 255) / 256), 256, 0, stream>>>(scc, att_out);
    }
    gemm(concat, p.wo2, p.bo2, nullptr, tmp, BT, D_, H_*D_, H_*D_, D_, D_, 0,0,0,1, 0, 0);
    k_ln_res<<<BT, 256, 0, stream>>>(x, tmp, p.ln2g, p.ln2b);

    // ---- FFN ----
    gemm(x, p.fw1, p.fb1, nullptr, hid, BT, 4*D_, D_, D_, 4*D_, 4*D_, 0,0,0,1, 0, 1);
    gemm(hid, p.fw2, p.fb2, nullptr, tmp, BT, D_, 4*D_, 4*D_, D_, D_, 0,0,0,1, 0, 0);
    k_ln_res<<<BT, 256, 0, stream>>>(x, tmp, p.ln3g, p.ln3b);
  }

  // final logits -> d_out (B,T,V+1) via padded weight copy (ldb = V1P)
  gemm(x, wpad, b_out, nullptr, y_out, BT, V1, D_, D_, V1P, V1, 0,0,0,1, 0, 0);
}
